// EchoStateNonlinNetwork_24550033064326
// MI455X (gfx1250) — matrix operand order repaired and checked by reading, not compiled, not run on hardware
//
#include <hip/hip_runtime.h>
#include <hip/hip_bf16.h>
#include <math.h>

typedef __attribute__((ext_vector_type(16))) __bf16 bf16x16;
typedef __attribute__((ext_vector_type(8)))  __bf16 bf16x8;
typedef __attribute__((ext_vector_type(8)))  float  f32x8;
typedef __attribute__((ext_vector_type(4)))  float  f32x4;
typedef unsigned int u32x4 __attribute__((ext_vector_type(4)));
typedef int          i32x8 __attribute__((ext_vector_type(8)));
typedef int          i32x4 __attribute__((ext_vector_type(4)));

#define B_  64
#define T_  512
#define E_  512
#define R_  1024
#define H_  1024
#define C_  50

#define KKRES (R_ / 32)        // 32 K-iterations for W_res
#define KKIN  (E_ / 32)        // 16 K-iterations for W_in
#define FRAG_ELEMS (32 * 16)   // 32 lanes x 16 bf16 per fragment
#define LDSW_BYTES ((4 * KKRES + 4 * KKIN) * FRAG_ELEMS * 2)  // 196608 (192KB)
#define XROW_BYTES 1056        // 512 bf16 (1024B) + 32B TDM pad per row
#define XBUF_BYTES (16 * XROW_BYTES)   // one 16x512 x-tile, padded

#if defined(__gfx1250__) && __has_builtin(__builtin_amdgcn_tensor_load_to_lds)
#define HAVE_TDM 1
#else
#define HAVE_TDM 0
#endif

// ---------------------------------------------------------------------------
// Fragment loaders (CDNA5 wave32 WMMA 16x16x32 bf16 layouts, CDNA5 ISA)
// ---------------------------------------------------------------------------

// A-matrix 16x32 (MxK), 16-bit: lane m(0..15) holds K = 0..7 (halves 0..7) and
// K = 16..23 (halves 8..15); lanes 16..31 hold the +8 K-offset variant.
__device__ __forceinline__ bf16x16 load_a_bf16(const __bf16* __restrict__ S,
                                               int m0, int kk, size_t ldk) {
  const int lane = threadIdx.x & 31;
  const int m    = lane & 15;
  const int kb   = (lane >> 4) << 3;   // 0 or 8
  const __bf16* p = S + (size_t)(m0 + m) * ldk + kk + kb;
  bf16x8 lo = *(const bf16x8*)(p);        // K = kb + 0..7   (16B aligned)
  bf16x8 hi = *(const bf16x8*)(p + 16);   // K = kb + 16..23
  bf16x16 a;
#pragma unroll
  for (int i = 0; i < 8; ++i) { a[i] = lo[i]; a[i + 8] = hi[i]; }
  return a;
}

// Same A layout, source fp32 -> convert to bf16 inline (fallback path).
__device__ __forceinline__ bf16x16 load_a_f32(const float* __restrict__ base,
                                              int m0, int kk, size_t row_stride) {
  const int lane = threadIdx.x & 31;
  const int m    = lane & 15;
  const int kb   = (lane >> 4) << 3;
  const float* p = base + (size_t)(m0 + m) * row_stride + kk + kb;
  f32x4 q0 = *(const f32x4*)(p);
  f32x4 q1 = *(const f32x4*)(p + 4);
  f32x4 q2 = *(const f32x4*)(p + 16);
  f32x4 q3 = *(const f32x4*)(p + 20);
  bf16x16 a;
#pragma unroll
  for (int i = 0; i < 4; ++i) {
    a[i]      = (__bf16)q0[i];
    a[i + 4]  = (__bf16)q1[i];
    a[i + 8]  = (__bf16)q2[i];
    a[i + 12] = (__bf16)q3[i];
  }
  return a;
}

// A fragment from the TDM-staged LDS x tile (16 rows, XROW_BYTES stride).
__device__ __forceinline__ bf16x16 load_a_ldsx(const char* xb, int ee) {
  const int lane = threadIdx.x & 31;
  const int m    = lane & 15;
  const int kb   = (lane >> 4) << 3;
  const char* p = xb + m * XROW_BYTES + (size_t)(ee + kb) * 2;
  bf16x8 lo = *(const bf16x8*)(p);
  bf16x8 hi = *(const bf16x8*)(p + 32);
  bf16x16 a;
#pragma unroll
  for (int i = 0; i < 8; ++i) { a[i] = lo[i]; a[i + 8] = hi[i]; }
  return a;
}

// B-matrix 32x16 (KxN): lane n(0..15) holds K=0..7 and K=16..23, lanes 16..31 hold the +8 K-offset variant.
// S @ W^T with row-major W means B[k,n] = W[n0+n, k]: two 16B chunks.
__device__ __forceinline__ bf16x16 load_b_global(const __bf16* __restrict__ W,
                                                 int n0, int kk, int ldk) {
  const int lane = threadIdx.x & 31;
  const int n    = lane & 15;
  const int ko   = (lane >> 4) << 3;   // 0 or 8
  const __bf16* p = W + (size_t)(n0 + n) * ldk + kk + ko;
  bf16x8 lo = *(const bf16x8*)(p);
  bf16x8 hi = *(const bf16x8*)(p + 16);
  bf16x16 b;
#pragma unroll
  for (int i = 0; i < 8; ++i) { b[i] = lo[i]; b[i + 8] = hi[i]; }
  return b;
}

// Fragment-major LDS layout: frag index -> 32 lanes x 32B contiguous.
__device__ __forceinline__ bf16x16 load_b_lds(const __bf16* lds, int frag) {
  const int lane = threadIdx.x & 31;
  return *(const bf16x16*)(lds + (((size_t)frag * 32 + lane) << 4));
}
__device__ __forceinline__ void store_b_lds(__bf16* lds, int frag, bf16x16 v) {
  const int lane = threadIdx.x & 31;
  *(bf16x16*)(lds + (((size_t)frag * 32 + lane) << 4)) = v;
}

// ---------------------------------------------------------------------------
// TDM: async 2D tile load xbf[m0:m0+16, t, :] -> LDS (CDNA5 ISA)
// ---------------------------------------------------------------------------
#if HAVE_TDM
__device__ __forceinline__ void tdm_load_x_tile(const __bf16* xbf, int t, int m0,
                                                unsigned ldsByteOff) {
  unsigned long long ga =
      (unsigned long long)(size_t)((const char*)xbf +
          ((size_t)m0 * ((size_t)T_ * E_) + (size_t)t * E_) * 2);
  u32x4 g0;
  g0[0] = 1u;                                  // count=1 (user descriptor)
  g0[1] = ldsByteOff;                          // lds_addr
  g0[2] = (unsigned)(ga & 0xffffffffu);        // global_addr[31:0]
  g0[3] = (unsigned)(ga >> 32) | (2u << 30);   // global_addr[56:32] | type=2
  i32x8 g1;
  g1[0] = (1 << 16)      // data_size = 2B
        | (1 << 20)      // pad_enable
        | (7 << 22)      // pad_interval: 256 dwords (= one 1024B row)
        | (7 << 25);     // pad_amount: 8 dwords (32B) -> 16B-aligned LDS rows
  g1[1] = (int)((unsigned)E_ << 16);                 // tensor_dim0[15:0] = 512
  g1[2] = (int)(((unsigned)(B_ * T_) & 0xffffu) << 16); // tensor_dim1[15:0] = 32768
  g1[3] = (int)((unsigned)E_ << 16);                 // tile_dim0 = 512
  g1[4] = 16;                                        // tile_dim1 = 16 rows
  g1[5] = (int)(T_ * E_);                            // tensor_dim0_stride = 262144
  g1[6] = 0;
  g1[7] = 0;
  i32x4 gz4 = {0, 0, 0, 0};                          // 2D tensor: groups 2/3 zero
  i32x8 gz8 = {0, 0, 0, 0, 0, 0, 0, 0};              // unused trailing group
  __builtin_amdgcn_tensor_load_to_lds(g0, g1, gz4, gz4, gz8, 0);
}
#endif

// ---------------------------------------------------------------------------
// Device-wide barrier (persistent kernel step sync)
// ---------------------------------------------------------------------------
__device__ __forceinline__ void grid_sync(unsigned* cnt, unsigned* gen, unsigned nblk) {
  __threadfence();
  __syncthreads();
  if (threadIdx.x == 0) {
    unsigned g = __hip_atomic_load(gen, __ATOMIC_RELAXED, __HIP_MEMORY_SCOPE_AGENT);
    unsigned arrived = __hip_atomic_fetch_add(cnt, 1u, __ATOMIC_ACQ_REL,
                                              __HIP_MEMORY_SCOPE_AGENT);
    if (arrived == nblk - 1) {
      __hip_atomic_store(cnt, 0u, __ATOMIC_RELAXED, __HIP_MEMORY_SCOPE_AGENT);
      __hip_atomic_fetch_add(gen, 1u, __ATOMIC_RELEASE, __HIP_MEMORY_SCOPE_AGENT);
    } else {
      while (__hip_atomic_load(gen, __ATOMIC_ACQUIRE, __HIP_MEMORY_SCOPE_AGENT) == g) {
        __builtin_amdgcn_s_sleep(1);
      }
    }
  }
  __syncthreads();
}

// ---------------------------------------------------------------------------
// Setup kernels
// ---------------------------------------------------------------------------
__global__ void esn_init(unsigned* bar, __bf16* S0, __bf16* W2pad) {
  int i = blockIdx.x * blockDim.x + threadIdx.x;
  if (i < 64) bar[i] = 0u;
  if (i < B_ * R_)  S0[i]    = (__bf16)0.0f;
  if (i < 64 * H_)  W2pad[i] = (__bf16)0.0f;
}

__global__ void cvt_f32_to_bf16(const float* __restrict__ src,
                                __bf16* __restrict__ dst, int n) {
  int i = blockIdx.x * blockDim.x + threadIdx.x;
  if (i < n) dst[i] = (__bf16)src[i];
}

// ---------------------------------------------------------------------------
// Persistent recurrence: state_t = tanh(state_{t-1} W_res^T + x_t W_in^T)
// 64 blocks x 128 threads = 256 waves; one 16x16 output tile per wave.
// W_res/W_in slices live in LDS (pre-swizzled B fragments) for all 512 steps.
// XMODE: 0 = x fp32 from global (cvt inline), 2 = x bf16 via TDM double buffer
//        (degrades to plain bf16 global loads if the TDM builtin is absent).
// ---------------------------------------------------------------------------
template <int XMODE>
__global__ void esn_recurrence(const float*  __restrict__ x,      // [B,T,E] f32
                               const __bf16* __restrict__ xbf,    // [B,T,E] bf16
                               const __bf16* __restrict__ Wres,   // [R,R] bf16
                               const __bf16* __restrict__ Win,    // [R,E] bf16
                               __bf16* S0, __bf16* S1,
                               unsigned* cnt, unsigned* gen) {
  extern __shared__ __align__(64) char smem[];
  __bf16* ldsWres = (__bf16*)smem;                                   // 4*32 frags
  __bf16* ldsWin  = (__bf16*)(smem + 4 * KKRES * FRAG_ELEMS * 2);    // 4*16 frags

  const int w    = threadIdx.x >> 5;          // wave in block: 0..3
  const int b    = blockIdx.x;                // 0..63
  const int m0   = (b >> 4) << 4;             // batch strip
  const int n0   = (((b & 15) << 2) + w) << 4;// reservoir strip
  const int lane = threadIdx.x & 31;
  const int hi   = lane >> 4;
  const int col  = lane & 15;

  // ---- stage loop-invariant weights into LDS, pre-swizzled as B fragments
#pragma unroll 4
  for (int kkIdx = 0; kkIdx < KKRES; ++kkIdx)
    store_b_lds(ldsWres, w * KKRES + kkIdx, load_b_global(Wres, n0, kkIdx * 32, R_));
#pragma unroll 4
  for (int eeIdx = 0; eeIdx < KKIN; ++eeIdx)
    store_b_lds(ldsWin, w * KKIN + eeIdx, load_b_global(Win, n0, eeIdx * 32, E_));
  __syncthreads();

#if HAVE_TDM
  if constexpr (XMODE == 2) {                 // prefetch x tile for t=0
    if (w == 0)
      tdm_load_x_tile(xbf, 0, m0, (unsigned)(size_t)(smem + LDSW_BYTES));
  }
#endif

#pragma unroll 1
  for (int t = 0; t < T_; ++t) {
    const __bf16* Scur = (t & 1) ? S1 : S0;
    __bf16*       Snxt = (t & 1) ? S0 : S1;

#if HAVE_TDM
    if constexpr (XMODE == 2) {               // async prefetch x tile for t+1
      if (w == 0 && t + 1 < T_)
        tdm_load_x_tile(xbf, t + 1, m0,
            (unsigned)(size_t)(smem + LDSW_BYTES + ((t + 1) & 1) * XBUF_BYTES));
    }
#endif

    f32x8 acc0 = {}, acc1 = {};   // two chains for WMMA ILP

    // state_{t-1} @ W_res^T  (K = 1024), weights from LDS
#pragma unroll 4
    for (int kkIdx = 0; kkIdx < KKRES; kkIdx += 2) {
      bf16x16 a0 = load_a_bf16(Scur, m0, kkIdx * 32, R_);
      bf16x16 b0 = load_b_lds(ldsWres, w * KKRES + kkIdx);
      acc0 = __builtin_amdgcn_wmma_f32_16x16x32_bf16(false, a0, false, b0,
                                                     (short)0, acc0, false, false);
      bf16x16 a1 = load_a_bf16(Scur, m0, (kkIdx + 1) * 32, R_);
      bf16x16 b1 = load_b_lds(ldsWres, w * KKRES + kkIdx + 1);
      acc1 = __builtin_amdgcn_wmma_f32_16x16x32_bf16(false, a1, false, b1,
                                                     (short)0, acc1, false, false);
    }

#if HAVE_TDM
    if constexpr (XMODE == 2) {               // x tile for step t is ready?
      if (t + 1 < T_) { if (w == 0) __builtin_amdgcn_s_wait_tensorcnt(1); }
      else            { if (w == 0) __builtin_amdgcn_s_wait_tensorcnt(0); }
      __syncthreads();
    }
#endif

    // x_t @ W_in^T  (K = 512)
    const float*  xrowf = x   + (size_t)t * E_;
    const __bf16* xrowb = xbf + (size_t)t * E_;
    const char*   xb    = smem + LDSW_BYTES + (t & 1) * XBUF_BYTES;
    (void)xrowf; (void)xrowb; (void)xb;
#pragma unroll 4
    for (int eeIdx = 0; eeIdx < KKIN; eeIdx += 2) {
      bf16x16 a0, a1;
      if constexpr (XMODE == 0) {
        a0 = load_a_f32(xrowf, m0, eeIdx * 32, (size_t)T_ * E_);
        a1 = load_a_f32(xrowf, m0, (eeIdx + 1) * 32, (size_t)T_ * E_);
      } else {
#if HAVE_TDM
        a0 = load_a_ldsx(xb, eeIdx * 32);
        a1 = load_a_ldsx(xb, (eeIdx + 1) * 32);
#else
        a0 = load_a_bf16(xrowb, m0, eeIdx * 32, (size_t)T_ * E_);
        a1 = load_a_bf16(xrowb, m0, (eeIdx + 1) * 32, (size_t)T_ * E_);
#endif
      }
      bf16x16 b0 = load_b_lds(ldsWin, w * KKIN + eeIdx);
      acc0 = __builtin_amdgcn_wmma_f32_16x16x32_bf16(false, a0, false, b0,
                                                     (short)0, acc0, false, false);
      bf16x16 b1 = load_b_lds(ldsWin, w * KKIN + eeIdx + 1);
      acc1 = __builtin_amdgcn_wmma_f32_16x16x32_bf16(false, a1, false, b1,
                                                     (short)0, acc1, false, false);
    }

    // D layout: VGPR v -> row m0 + v + 8*hi, col n0 + (lane&15)
#pragma unroll
    for (int v = 0; v < 8; ++v) {
      int row = m0 + v + (hi << 3);
      Snxt[(size_t)row * R_ + n0 + col] = (__bf16)tanhf(acc0[v] + acc1[v]);
    }
    grid_sync(cnt, gen, gridDim.x);
  }
  // t=511 (odd) wrote S0 -> final state lives in S0
}

// ---------------------------------------------------------------------------
// Classifier: h = relu(S W1^T + b1)   [64x1024], 256 tiles
// ---------------------------------------------------------------------------
__global__ void esn_classifier_h(const __bf16* __restrict__ S,
                                 const __bf16* __restrict__ W1b,
                                 const float*  __restrict__ b1,
                                 __bf16* __restrict__ h) {
  const int wave = blockIdx.x * (blockDim.x >> 5) + (threadIdx.x >> 5);
  const int m0 = (wave >> 6) << 4, n0 = (wave & 63) << 4;
  const int lane = threadIdx.x & 31;
  const int hi = lane >> 4, col = lane & 15;
  f32x8 acc = {};
#pragma unroll 4
  for (int kk = 0; kk < R_; kk += 32) {
    bf16x16 a = load_a_bf16(S, m0, kk, R_);
    bf16x16 b = load_b_global(W1b, n0, kk, R_);
    acc = __builtin_amdgcn_wmma_f32_16x16x32_bf16(false, a, false, b,
                                                  (short)0, acc, false, false);
  }
  const float bias = b1[n0 + col];
#pragma unroll
  for (int v = 0; v < 8; ++v) {
    int row = m0 + v + (hi << 3);
    h[(size_t)row * H_ + n0 + col] = (__bf16)fmaxf(acc[v] + bias, 0.0f);
  }
}

// logits = h W2^T + b2   [64x50]; W2 zero-padded to 64 rows -> 16 tiles (2 blocks)
__global__ void esn_classifier_out(const __bf16* __restrict__ h,
                                   const __bf16* __restrict__ W2p,
                                   const float*  __restrict__ b2,
                                   float* __restrict__ out) {
  const int wave = blockIdx.x * (blockDim.x >> 5) + (threadIdx.x >> 5);
  const int m0 = (wave >> 2) << 4, n0 = (wave & 3) << 4;
  const int lane = threadIdx.x & 31;
  const int hi = lane >> 4, col = lane & 15;
  f32x8 acc = {};
#pragma unroll 4
  for (int kk = 0; kk < H_; kk += 32) {
    bf16x16 a = load_a_bf16(h, m0, kk, H_);
    bf16x16 b = load_b_global(W2p, n0, kk, H_);
    acc = __builtin_amdgcn_wmma_f32_16x16x32_bf16(false, a, false, b,
                                                  (short)0, acc, false, false);
  }
  const int cg = n0 + col;
  if (cg < C_) {
    const float bias = b2[cg];
#pragma unroll
    for (int v = 0; v < 8; ++v) {
      int row = m0 + v + (hi << 3);
      out[row * C_ + cg] = acc[v] + bias;
    }
  }
}

// ---------------------------------------------------------------------------
extern "C" void kernel_launch(void* const* d_in, const int* in_sizes, int n_in,
                              void* d_out, int out_size, void* d_ws, size_t ws_size,
                              hipStream_t stream) {
  const float* x     = (const float*)d_in[0];
  const float* W_in  = (const float*)d_in[1];
  const float* W_res = (const float*)d_in[2];
  const float* W1    = (const float*)d_in[3];
  const float* b1    = (const float*)d_in[4];
  const float* W2    = (const float*)d_in[5];
  const float* b2    = (const float*)d_in[6];

  // Workspace layout (bytes), all 256-aligned
  char* ws = (char*)d_ws;
  unsigned* bar   = (unsigned*)ws;                         size_t off = 256;
  __bf16* S0      = (__bf16*)(ws + off); off += (size_t)B_ * R_ * 2;
  __bf16* S1      = (__bf16*)(ws + off); off += (size_t)B_ * R_ * 2;
  __bf16* Wres_b  = (__bf16*)(ws + off); off += (size_t)R_ * R_ * 2;
  __bf16* Win_b   = (__bf16*)(ws + off); off += (size_t)R_ * E_ * 2;
  __bf16* W1_b    = (__bf16*)(ws + off); off += (size_t)H_ * R_ * 2;
  __bf16* W2_b    = (__bf16*)(ws + off); off += (size_t)64 * H_ * 2;  // padded
  __bf16* hbuf    = (__bf16*)(ws + off); off += (size_t)B_ * H_ * 2;
  __bf16* xbf     = (__bf16*)(ws + off);
  const size_t xbfBytes = (size_t)B_ * T_ * E_ * 2;        // 33.5 MB
  const bool have_xbf = (ws_size >= off + xbfBytes);

  esn_init<<<256, 256, 0, stream>>>(bar, S0, W2_b);

  cvt_f32_to_bf16<<<(R_ * R_ + 255) / 256, 256, 0, stream>>>(W_res, Wres_b, R_ * R_);
  cvt_f32_to_bf16<<<(R_ * E_ + 255) / 256, 256, 0, stream>>>(W_in, Win_b, R_ * E_);
  cvt_f32_to_bf16<<<(H_ * R_ + 255) / 256, 256, 0, stream>>>(W1, W1_b, H_ * R_);
  cvt_f32_to_bf16<<<(C_ * H_ + 255) / 256, 256, 0, stream>>>(W2, W2_b, C_ * H_);

  if (have_xbf) {
    cvt_f32_to_bf16<<<(B_ * T_ * E_ + 255) / 256, 256, 0, stream>>>(x, xbf, B_ * T_ * E_);
    esn_recurrence<2><<<64, 128, LDSW_BYTES + 2 * XBUF_BYTES, stream>>>(
        x, xbf, Wres_b, Win_b, S0, S1, bar, bar + 1);
  } else {
    esn_recurrence<0><<<64, 128, LDSW_BYTES, stream>>>(
        x, xbf, Wres_b, Win_b, S0, S1, bar, bar + 1);
  }

  esn_classifier_h<<<32, 256, 0, stream>>>(S0, W1_b, b1, hbuf);
  esn_classifier_out<<<2, 256, 0, stream>>>(hbuf, W2_b, b2, (float*)d_out);
}
